// AttentiveRNNLanguageModel_65429531788044
// MI455X (gfx1250) — compile-verified, matrix operand order checked by reading, not run on hardware
//
#include <hip/hip_runtime.h>

// ---------- types ----------
typedef __attribute__((ext_vector_type(16))) __bf16   v16bf;
typedef __attribute__((ext_vector_type(8)))  float    v8f;
typedef __attribute__((ext_vector_type(4)))  unsigned uint4v;
typedef __attribute__((ext_vector_type(4)))  int      int4v;

typedef unsigned short ushort_t;

// ---------- bf16 helpers ----------
__device__ __forceinline__ ushort_t f2bf(float f) {
    unsigned u = __float_as_uint(f);
    u += 0x7FFFu + ((u >> 16) & 1u);          // round-to-nearest-even
    return (ushort_t)(u >> 16);
}
__device__ __forceinline__ float bflo(unsigned u) { return __uint_as_float(u << 16); }
__device__ __forceinline__ float bfhi(unsigned u) { return __uint_as_float(u & 0xFFFF0000u); }

// ---------- CDNA5 async copy helpers (ASYNCcnt path) ----------
// Builtin signature (from clang-22 diagnostic): first param is
// "int __vector(4) __device__ *" i.e. non-const v4i32 in addrspace(1);
// LDS side is the addrspace(3) counterpart.
typedef __attribute__((address_space(1))) int4v* g_b128_ptr;
typedef __attribute__((address_space(3))) int4v* l_b128_ptr;

__device__ __forceinline__ void async_load_b128(void* lptr, const void* gptr) {
#if __has_builtin(__builtin_amdgcn_global_load_async_to_lds_b128)
    __builtin_amdgcn_global_load_async_to_lds_b128(
        (g_b128_ptr)(void*)gptr, (l_b128_ptr)lptr, 0, 0);
#else
    // VDST = LDS byte address VGPR, then 64-bit global VGPR address, saddr=off
    unsigned l = (unsigned)(unsigned long long)
                 (__attribute__((address_space(3))) void*)lptr;
    asm volatile("global_load_async_to_lds_b128 %0, %1, off"
                 :: "v"(l), "v"(gptr) : "memory");
#endif
}

template <int N>
__device__ __forceinline__ void wait_asynccnt() {
#if __has_builtin(__builtin_amdgcn_s_wait_asynccnt)
    __builtin_amdgcn_s_wait_asynccnt((short)N);
#else
    asm volatile("s_wait_asynccnt %0" :: "n"(N) : "memory");
#endif
}

// ---------- model dims ----------
#define DV 32000
#define DE 1024
#define DH 1024
#define DB 4
#define DS 512
#define DM (DB*DS)        // 2048 rows of (b,s)

// =====================================================================
// fp32 -> bf16 convert (grid-stride)
// =====================================================================
__global__ void k_f32_to_bf16(const float* __restrict__ src,
                              ushort_t* __restrict__ dst, long n) {
    long i = (long)blockIdx.x * blockDim.x + threadIdx.x;
    long stride = (long)gridDim.x * blockDim.x;
    for (; i < n; i += stride) dst[i] = f2bf(src[i]);
}

// =====================================================================
// embedding gather -> bf16 x [2048,1024]
// =====================================================================
__global__ void k_embed(const int* __restrict__ ids,
                        const float* __restrict__ emb,
                        ushort_t* __restrict__ xbf) {
    long i = (long)blockIdx.x * blockDim.x + threadIdx.x; // over 2048*1024
    long row = i >> 10, e = i & 1023;
    xbf[i] = f2bf(emb[(long)ids[row] * DE + e]);
}

// =====================================================================
// bias_sum = b_ih + b_hh ; reset grid-sync counter
// =====================================================================
__global__ void k_bias_init(const float* __restrict__ bih,
                            const float* __restrict__ bhh,
                            float* __restrict__ bias,
                            unsigned* __restrict__ ctr) {
    int i = blockIdx.x * blockDim.x + threadIdx.x;
    if (i < 4 * DH) bias[i] = bih[i] + bhh[i];
    if (i == 0) *ctr = 0u;
}

// =====================================================================
// Generic bf16 WMMA GEMM:  C[M,N] = A[M,K] * B[N,K]^T (+bias[n]) (tanh?)
// block tile 128x128, BK=32, 256 threads = 8 waves in 2(M)x4(N) grid,
// each wave 64x32 = 4x2 v_wmma_f32_16x16x32_bf16 frags.
// Tile staging: double-buffered GLOBAL_LOAD_ASYNC_TO_LDS_B128 (ASYNCcnt);
// async loads retire in order, so waiting asynccnt<=4 with tile k+1 in
// flight guarantees tile k is resident.
// Requires M%128==0, N%128==0, K%32==0 (true for all uses here).
// =====================================================================
#define LDSK 40   // 32 + 8 pad (80B row stride -> conflict-free frag loads)

__global__ void __launch_bounds__(256)
k_gemm_bf16(const ushort_t* __restrict__ A, const ushort_t* __restrict__ B,
            const float* __restrict__ bias, float* __restrict__ C,
            int M, int N, int K, int act) {
    __shared__ __align__(16) ushort_t As[2][128 * LDSK];
    __shared__ __align__(16) ushort_t Bs[2][128 * LDSK];

    const int tid  = threadIdx.x;
    const int m0   = blockIdx.y * 128;
    const int n0   = blockIdx.x * 128;
    const int wave = tid >> 5, lane = tid & 31;
    const int wm   = wave >> 2, wn = wave & 3;     // 2 x 4 wave grid
    const int lr   = lane & 15, lhi = lane >> 4;

    // this thread's two 16B segments of the 128x32 tile (512 segs / 256 thr)
    const int r0 = tid >> 2,         q0 = tid & 3;
    const int r1 = (tid + 256) >> 2, q1 = (tid + 256) & 3;

    v8f acc[4][2];
#pragma unroll
    for (int mt = 0; mt < 4; ++mt)
#pragma unroll
        for (int nt = 0; nt < 2; ++nt)
#pragma unroll
            for (int r = 0; r < 8; ++r) acc[mt][nt][r] = 0.0f;

    union Frag { uint4v u[2]; v16bf v; };

    const int nk = K >> 5;

    // ---- stage tile 0 into buffer 0 (4 async b128 per thread) ----
    {
        const int k0 = 0;
        async_load_b128(&As[0][r0 * LDSK + q0 * 8],
                        A + (size_t)(m0 + r0) * K + k0 + q0 * 8);
        async_load_b128(&As[0][r1 * LDSK + q1 * 8],
                        A + (size_t)(m0 + r1) * K + k0 + q1 * 8);
        async_load_b128(&Bs[0][r0 * LDSK + q0 * 8],
                        B + (size_t)(n0 + r0) * K + k0 + q0 * 8);
        async_load_b128(&Bs[0][r1 * LDSK + q1 * 8],
                        B + (size_t)(n0 + r1) * K + k0 + q1 * 8);
    }

    for (int kk = 0; kk < nk; ++kk) {
        const int cur = kk & 1;
        if (kk + 1 < nk) {
            // ---- stage tile kk+1 into the other buffer ----
            const int nxt = cur ^ 1;
            const int k1  = (kk + 1) << 5;
            async_load_b128(&As[nxt][r0 * LDSK + q0 * 8],
                            A + (size_t)(m0 + r0) * K + k1 + q0 * 8);
            async_load_b128(&As[nxt][r1 * LDSK + q1 * 8],
                            A + (size_t)(m0 + r1) * K + k1 + q1 * 8);
            async_load_b128(&Bs[nxt][r0 * LDSK + q0 * 8],
                            B + (size_t)(n0 + r0) * K + k1 + q0 * 8);
            async_load_b128(&Bs[nxt][r1 * LDSK + q1 * 8],
                            B + (size_t)(n0 + r1) * K + k1 + q1 * 8);
            // speculative prefetch of the slab after that into GL2
            __builtin_prefetch(A + (size_t)(m0 + r0) * K + k1 + 32, 0, 1);
            __builtin_prefetch(B + (size_t)(n0 + r0) * K + k1 + 32, 0, 1);
            wait_asynccnt<4>();   // tile kk done; tile kk+1 may stay in flight
        } else {
            wait_asynccnt<0>();   // last tile: drain everything
        }
        __syncthreads();          // all waves' tile-kk segments resident

        // ---- fragment loads (2 x ds_load_b128 each) ----
        Frag a[4], b[2];
#pragma unroll
        for (int mt = 0; mt < 4; ++mt) {
            const ushort_t* p = &As[cur][(wm * 64 + mt * 16 + lr) * LDSK + lhi * 8];
            a[mt].u[0] = *(const uint4v*)p;
            a[mt].u[1] = *(const uint4v*)(p + 16);
        }
#pragma unroll
        for (int nt = 0; nt < 2; ++nt) {
            const ushort_t* p = &Bs[cur][(wn * 32 + nt * 16 + lr) * LDSK + lhi * 8];
            b[nt].u[0] = *(const uint4v*)p;
            b[nt].u[1] = *(const uint4v*)(p + 16);
        }
        // ---- 8 WMMAs ----
#pragma unroll
        for (int mt = 0; mt < 4; ++mt)
#pragma unroll
            for (int nt = 0; nt < 2; ++nt)
                acc[mt][nt] = __builtin_amdgcn_wmma_f32_16x16x32_bf16(
                    false, a[mt].v, false, b[nt].v, (short)0, acc[mt][nt],
                    false, false);
        __syncthreads();          // done reading buf cur before it is refilled
    }

    // ---- epilogue: C layout lane(l<16)->M=r, lane(l>=16)->M=r+8, N=l&15 ----
#pragma unroll
    for (int mt = 0; mt < 4; ++mt)
#pragma unroll
        for (int nt = 0; nt < 2; ++nt) {
            const int rowb = m0 + wm * 64 + mt * 16 + lhi * 8;
            const int col  = n0 + wn * 32 + nt * 16 + lr;
            const float bv = bias ? bias[col] : 0.0f;
#pragma unroll
            for (int r = 0; r < 8; ++r) {
                float v = acc[mt][nt][r] + bv;
                if (act) v = tanhf(v);
                C[(size_t)(rowb + r) * N + col] = v;
            }
        }
}

// =====================================================================
// LSTM scan: persistent kernel, 128 blocks x 256 threads, grid barrier
// per step. Block owns 8 H-columns (32 gate rows x 4 batch), split-K=2.
// =====================================================================
__global__ void __launch_bounds__(256)
k_lstm(const ushort_t* __restrict__ Whh,   // [4096,1024] bf16
       const float*    __restrict__ xg,    // [2048,4096] fp32 precomputed gates
       ushort_t*       __restrict__ hs_bf, // [B,S,H] bf16 out
       float*          __restrict__ hs_f,  // [B,S,H] fp32 out
       unsigned*       __restrict__ ctr) {
    __shared__ __align__(16) ushort_t h_lds[DB * DH]; // prev h, bf16
    __shared__ float part[256];
    __shared__ float gates[32 * 4];

    const int tid   = threadIdx.x;
    const unsigned G = gridDim.x;
    const int jbase = blockIdx.x * 8;
    const int ks    = tid >> 7;        // K split 0/1
    const int rb    = tid & 127;
    const int bb    = rb & 3;          // batch
    const int row   = rb >> 2;         // 0..31 = gate*8 + jj
    const int g     = row >> 3, jj = row & 7;
    const int wrow  = g * DH + jbase + jj;   // global W_hh row

    float c = 0.0f;                    // cell state owned by tid<32

    for (int s = 0; s < DS; ++s) {
        // ---- stage h_{t-1} (all batches) into LDS ----
        for (int p = tid; p < 512; p += 256) {
            const int bl = p >> 7, j8 = p & 127;
            uint4v v;
            if (s == 0) { v.x = v.y = v.z = v.w = 0u; }
            else v = *(const uint4v*)(hs_bf + ((size_t)(bl * DS + s - 1)) * DH + j8 * 8);
            *(uint4v*)&h_lds[bl * DH + j8 * 8] = v;
        }
        __syncthreads();

        // ---- partial dot: 512 MACs/thread over packed bf16 pairs ----
        const uint4v* wp = (const uint4v*)(Whh + (size_t)wrow * DH + ks * 512);
        const uint4v* hp = (const uint4v*)&h_lds[bb * DH + ks * 512];
        float acc = 0.0f;
#pragma unroll 4
        for (int it = 0; it < 64; ++it) {
            uint4v w = wp[it], h = hp[it];
            acc += bflo(w.x) * bflo(h.x) + bfhi(w.x) * bfhi(h.x);
            acc += bflo(w.y) * bflo(h.y) + bfhi(w.y) * bfhi(h.y);
            acc += bflo(w.z) * bflo(h.z) + bfhi(w.z) * bfhi(h.z);
            acc += bflo(w.w) * bflo(h.w) + bfhi(w.w) * bfhi(h.w);
        }
        part[tid] = acc;
        __syncthreads();

        // ---- reduce split-K, add precomputed input gates ----
        if (tid < 128) {
            const float sum = part[tid] + part[tid + 128];
            gates[row * 4 + bb] =
                xg[((size_t)(bb * DS + s)) * (4 * DH) + wrow] + sum;
        }
        __syncthreads();

        // ---- cell update: 32 threads own (jj, b) pairs ----
        if (tid < 32) {
            const int j2 = tid >> 2, b2 = tid & 3;
            const float gi = gates[(0  + j2) * 4 + b2];
            const float gf = gates[(8  + j2) * 4 + b2];
            const float gg = gates[(16 + j2) * 4 + b2];
            const float go = gates[(24 + j2) * 4 + b2];
            const float i_ = 1.0f / (1.0f + expf(-gi));
            const float f_ = 1.0f / (1.0f + expf(-gf));
            const float g_ = tanhf(gg);
            const float o_ = 1.0f / (1.0f + expf(-go));
            c = f_ * c + i_ * g_;
            const float h_ = o_ * tanhf(c);
            const size_t idx = ((size_t)(b2 * DS + s)) * DH + jbase + j2;
            hs_f[idx]  = h_;
            hs_bf[idx] = f2bf(h_);
        }
        __threadfence();
        __syncthreads();
        // ---- grid-wide barrier ----
        if (tid == 0) {
            atomicAdd(ctr, 1u);
            const unsigned target = (unsigned)(s + 1) * G;
            while (__hip_atomic_load(ctr, __ATOMIC_ACQUIRE,
                                     __HIP_MEMORY_SCOPE_AGENT) < target) {
                __builtin_amdgcn_s_sleep(1);
            }
        }
        __syncthreads();
    }
}

// =====================================================================
// attention scores: sc[row] = tanh_t[row,:] . attn2_W + attn2_b
// =====================================================================
__global__ void __launch_bounds__(128)
k_scores(const float* __restrict__ tt, const float* __restrict__ w2,
         const float* __restrict__ b2, float* __restrict__ sc) {
    __shared__ float red[128];
    const int row = blockIdx.x, tid = threadIdx.x;
    float a = 0.0f;
    for (int j = tid; j < DH; j += 128) a += tt[(size_t)row * DH + j] * w2[j];
    red[tid] = a;
    __syncthreads();
    for (int st = 64; st > 0; st >>= 1) {
        if (tid < st) red[tid] += red[tid + st];
        __syncthreads();
    }
    if (tid == 0) sc[row] = red[0] + b2[0];
}

// =====================================================================
// causal softmax pooling + concat: writes combined[2048,2048] bf16
// combined[b*S+s] = [ cumsum(e*h)/cumsum(e) , h ]
// one block per batch, 1024 threads (thread = H column)
// =====================================================================
__global__ void __launch_bounds__(1024)
k_pool(const float* __restrict__ sc, const float* __restrict__ hs_f,
       ushort_t* __restrict__ comb) {
    __shared__ float e[DS];
    __shared__ float den[DS];
    __shared__ float red[DS];
    const int b = blockIdx.x, tid = threadIdx.x;

    if (tid < DS) red[tid] = sc[b * DS + tid];
    __syncthreads();
    for (int st = 256; st > 0; st >>= 1) {
        if (tid < st) red[tid] = fmaxf(red[tid], red[tid + st]);
        __syncthreads();
    }
    const float mx = red[0];
    if (tid < DS) e[tid] = expf(sc[b * DS + tid] - mx);
    __syncthreads();
    if (tid == 0) {
        float r = 0.0f;
        for (int s = 0; s < DS; ++s) { r += e[s]; den[s] = r; }
    }
    __syncthreads();

    const int j = tid;            // H column
    float num = 0.0f;
    for (int s = 0; s < DS; ++s) {
        const float hv = hs_f[((size_t)(b * DS + s)) * DH + j];
        num += e[s] * hv;
        const float ctx = num / den[s];
        const size_t base = ((size_t)(b * DS + s)) * (2 * DH);
        comb[base + j]      = f2bf(ctx);
        comb[base + DH + j] = f2bf(hv);
    }
}

// =====================================================================
// launcher
// =====================================================================
extern "C" void kernel_launch(void* const* d_in, const int* in_sizes, int n_in,
                              void* d_out, int out_size, void* d_ws, size_t ws_size,
                              hipStream_t stream) {
    const int*   ids     = (const int*)  d_in[0];
    const float* emb     = (const float*)d_in[1];
    const float* W_ih    = (const float*)d_in[2];
    const float* W_hh    = (const float*)d_in[3];
    const float* b_ih    = (const float*)d_in[4];
    const float* b_hh    = (const float*)d_in[5];
    const float* attn1_W = (const float*)d_in[6];
    const float* attn1_b = (const float*)d_in[7];
    const float* attn2_W = (const float*)d_in[8];
    const float* attn2_b = (const float*)d_in[9];
    const float* dec_W   = (const float*)d_in[10];
    const float* dec_b   = (const float*)d_in[11];
    float* out = (float*)d_out;

    char* ws = (char*)d_ws;
    size_t off = 0;
    auto alloc = [&](size_t bytes) -> char* {
        char* p = ws + off;
        off += (bytes + 255) & ~(size_t)255;
        return p;
    };
    ushort_t* x_bf    = (ushort_t*)alloc((size_t)DM * DE * 2);
    ushort_t* Wih_bf  = (ushort_t*)alloc((size_t)4 * DH * DE * 2);
    ushort_t* Whh_bf  = (ushort_t*)alloc((size_t)4 * DH * DH * 2);
    ushort_t* A1_bf   = (ushort_t*)alloc((size_t)DH * DH * 2);
    ushort_t* dec_bf  = (ushort_t*)alloc((size_t)DV * 2 * DH * 2);
    float*    xg      = (float*)   alloc((size_t)DM * 4 * DH * 4);
    float*    hs_f    = (float*)   alloc((size_t)DM * DH * 4);
    ushort_t* hs_bf   = (ushort_t*)alloc((size_t)DM * DH * 2);
    float*    tanh_t  = (float*)   alloc((size_t)DM * DH * 4);
    float*    sc      = (float*)   alloc((size_t)DM * 4);
    ushort_t* comb    = (ushort_t*)alloc((size_t)DM * 2 * DH * 2);
    float*    biassum = (float*)   alloc((size_t)4 * DH * 4);
    unsigned* ctr     = (unsigned*)alloc(256);

    // weight conversions (fp32 -> bf16; dec_W: 262MB -> 131MB, L2-resident)
    k_f32_to_bf16<<<2048, 256, 0, stream>>>(W_ih,    Wih_bf, (long)4 * DH * DE);
    k_f32_to_bf16<<<2048, 256, 0, stream>>>(W_hh,    Whh_bf, (long)4 * DH * DH);
    k_f32_to_bf16<<<2048, 256, 0, stream>>>(attn1_W, A1_bf,  (long)DH * DH);
    k_f32_to_bf16<<<8192, 256, 0, stream>>>(dec_W,   dec_bf, (long)DV * 2 * DH);
    k_bias_init<<<16, 256, 0, stream>>>(b_ih, b_hh, biassum, ctr);

    // x = emb[ids] -> bf16
    k_embed<<<(DM * DE) / 256, 256, 0, stream>>>(ids, emb, x_bf);

    // xg = x . W_ih^T + (b_ih + b_hh)      [2048 x 4096], K=1024
    k_gemm_bf16<<<dim3((4 * DH) / 128, DM / 128), 256, 0, stream>>>(
        x_bf, Wih_bf, biassum, xg, DM, 4 * DH, DE, 0);

    // sequential LSTM scan (persistent, grid-synced per step)
    k_lstm<<<128, 256, 0, stream>>>(Whh_bf, xg, hs_bf, hs_f, ctr);

    // tanh(h . attn1_W^T + attn1_b)        [2048 x 1024], K=1024, fused tanh
    k_gemm_bf16<<<dim3(DH / 128, DM / 128), 256, 0, stream>>>(
        hs_bf, A1_bf, attn1_b, tanh_t, DM, DH, DH, 1);

    // scores + causal softmax pooling -> combined bf16 [2048 x 2048]
    k_scores<<<DM, 128, 0, stream>>>(tanh_t, attn2_W, attn2_b, sc);
    k_pool<<<DB, 1024, 0, stream>>>(sc, hs_f, comb);

    // logits = combined . dec_W^T + dec_b  [2048 x 32000], K=2048
    k_gemm_bf16<<<dim3(DV / 128, DM / 128), 256, 0, stream>>>(
        comb, dec_bf, dec_b, out, DM, DV, 2 * DH, 0);
}
